// FPMamba2_48919677501476
// MI455X (gfx1250) — matrix operand order repaired and checked by reading, not compiled, not run on hardware
//
#include <hip/hip_runtime.h>
#include <hip/hip_bf16.h>

// ---------------------------------------------------------------------------
// Problem constants (from the reference)
// ---------------------------------------------------------------------------
#define B_SZ   2
#define L_SZ   2048
#define DM_SZ  1024
#define DI_SZ  2048
#define DS_SZ  128
#define NH_SZ  32
#define HP_SZ  64
#define TOK    (B_SZ * L_SZ)               // 4096 tokens
#define DIP    (2*DI_SZ + 2*DS_SZ + NH_SZ) // 4384
#define DIP_PAD 4480                       // 4384 padded up to a multiple of 128
#define PDIM   (64*64 + 32*32)             // 5120
#define N_ITER 4                           // N_FP + N_BW
#define EPS_F  1e-5f

typedef __bf16 bf16;
typedef __attribute__((ext_vector_type(16))) __bf16 bf16x16;
typedef __attribute__((ext_vector_type(8)))  __bf16 bf16x8;
typedef __attribute__((ext_vector_type(8)))  float  f32x8;
typedef int v4i __attribute__((vector_size(16)));   // matches builtin param type

#if __has_builtin(__builtin_amdgcn_global_load_async_to_lds_b128) && \
    __has_builtin(__builtin_amdgcn_s_wait_asynccnt)
#define USE_ASYNC_LDS 1
#else
#define USE_ASYNC_LDS 0
#endif

__device__ __forceinline__ bf16 f32_to_bf16(float f) {
    unsigned u = __builtin_bit_cast(unsigned, f);
    unsigned r = u + 0x7FFFu + ((u >> 16) & 1u);   // round-to-nearest-even
    unsigned short h = (unsigned short)(r >> 16);
    return __builtin_bit_cast(bf16, h);
}
__device__ __forceinline__ bf16 bf16_zero() {
    return __builtin_bit_cast(bf16, (unsigned short)0);
}
__device__ __forceinline__ float siluf(float v) { return v / (1.f + __expf(-v)); }

// ---------------------------------------------------------------------------
// WMMA fragment loaders (contiguous k order chosen by this code, same for A and B)
// A 16x32 (MxK): lanes 0-15 -> M=lane, K = 0..15 contiguous
//                lanes 16-31 -> M=lane-16, K = 16..31 contiguous
// B 32x16 (KxN): lanes 0-15 -> N=lane, K = 0..15 contiguous
//                lanes 16-31 -> N=lane-16, K = 16..31 contiguous
// ---------------------------------------------------------------------------
__device__ __forceinline__ bf16x16 load_a_frag(const bf16* A, int lda, int m0, int k0, int lane) {
    int m    = m0 + (lane & 15);
    int koff = (lane >> 4) ? 16 : 0;
    const bf16* row = A + (size_t)m * lda + k0 + koff;
    bf16x8 lo = *(const bf16x8*)(row);
    bf16x8 hi = *(const bf16x8*)(row + 8);
    bf16x16 f;
#pragma unroll
    for (int i = 0; i < 8; ++i) { f[i] = lo[i]; f[8 + i] = hi[i]; }
    return f;
}
__device__ __forceinline__ bf16x16 load_b_frag(const bf16* B, int ldb, int n0, int k0, int lane) {
    int n    = n0 + (lane & 15);
    int koff = (lane >> 4) ? 16 : 0;
    return *(const bf16x16*)(B + (size_t)n * ldb + k0 + koff);
}
__device__ __forceinline__ f32x8 wmma_bf16(bf16x16 a, bf16x16 b, f32x8 c) {
    return __builtin_amdgcn_wmma_f32_16x16x32_bf16(false, a, false, b, (short)0, c, false, false);
}

// ---------------------------------------------------------------------------
// LDS-tiled GEMM: C(MxN,f32) = A(MxK,bf16) @ B(NxK,bf16)^T
// Block = 256 threads (8 waves). Block tile BM=BN=128, BK=32, double-buffered
// LDS staging via async load-to-LDS (ASYNCcnt) when available.
// Each wave owns a 32x64 register tile -> 8 WMMAs per K stage.
// Requires: M % 128 == 0, N % 128 == 0, K % 32 == 0.
// ---------------------------------------------------------------------------
#define BM 128
#define BN 128
#define BK 32
#define LDT 48   // padded LDS row stride (elements); 96B rows -> 32B-aligned frags

__device__ __forceinline__ void stage_tile(
    const bf16* __restrict__ G, int ld, int row0, int k0,
    bf16* lds, int tid)
{
    // tile = 128 rows x 32 cols = 512 chunks of 8 bf16; 2 chunks per thread
#pragma unroll
    for (int i = 0; i < 2; ++i) {
        int c   = tid * 2 + i;
        int row = c >> 2;            // 4 chunks per row
        int col = (c & 3) * 8;
        const bf16* src = G + (size_t)(row0 + row) * ld + k0 + col;
        bf16* dst = lds + row * LDT + col;
#if USE_ASYNC_LDS
        __builtin_amdgcn_global_load_async_to_lds_b128(
            (__attribute__((address_space(1))) v4i*)src,
            (__attribute__((address_space(3))) v4i*)dst, 0, 0);
#else
        *(uint4*)dst = *(const uint4*)src;
#endif
    }
}

__global__ __launch_bounds__(256) void gemm_bf16_lds(
    const bf16* __restrict__ A, const bf16* __restrict__ B, float* __restrict__ C,
    int M, int N, int K, int lda, int ldb, int ldc)
{
    __shared__ __align__(32) bf16 As[2][BM * LDT];
    __shared__ __align__(32) bf16 Bs[2][BN * LDT];

    const int tid  = threadIdx.x;
    const int lane = tid & 31;
    const int wave = tid >> 5;
    const int bn_tiles = N / BN;
    const int bm = (blockIdx.x / bn_tiles) * BM;
    const int bn = (blockIdx.x % bn_tiles) * BN;
    const int wm = (wave >> 1) * 32;  // wave row offset within block tile
    const int wn = (wave & 1) * 64;   // wave col offset within block tile

    f32x8 acc[2][4] = {};

    stage_tile(A, lda, bm, 0, As[0], tid);
    stage_tile(B, ldb, bn, 0, Bs[0], tid);

    const int nk = K / BK;
    for (int ks = 0; ks < nk; ++ks) {
        const int cur = ks & 1, nxt = cur ^ 1;
        if (ks + 1 < nk) {
            if (ks + 2 < nk) { // L2 warm-up hint two stages ahead
                __builtin_prefetch(A + (size_t)(bm + (tid >> 1)) * lda + (ks + 2) * BK, 0, 1);
                __builtin_prefetch(B + (size_t)(bn + (tid >> 1)) * ldb + (ks + 2) * BK, 0, 1);
            }
            stage_tile(A, lda, bm, (ks + 1) * BK, As[nxt], tid);
            stage_tile(B, ldb, bn, (ks + 1) * BK, Bs[nxt], tid);
#if USE_ASYNC_LDS
            __builtin_amdgcn_s_wait_asynccnt(4);  // stage ks fully landed
#endif
        } else {
#if USE_ASYNC_LDS
            __builtin_amdgcn_s_wait_asynccnt(0);
#endif
        }
        __syncthreads();

        bf16x16 a0 = load_a_frag(As[cur], LDT, wm,      0, lane);
        bf16x16 a1 = load_a_frag(As[cur], LDT, wm + 16, 0, lane);
#pragma unroll
        for (int j = 0; j < 4; ++j) {
            bf16x16 b = load_b_frag(Bs[cur], LDT, wn + j * 16, 0, lane);
            acc[0][j] = wmma_bf16(a0, b, acc[0][j]);
            acc[1][j] = wmma_bf16(a1, b, acc[1][j]);
        }
        __syncthreads();   // frees buf[cur] for the stage issued next iteration
    }

    // C/D layout: VGPR r, lane l -> M = r + 8*(l>>4), N = l&15
    const int cn = lane & 15, ro = (lane >> 4) * 8;
#pragma unroll
    for (int i = 0; i < 2; ++i)
#pragma unroll
        for (int j = 0; j < 4; ++j)
#pragma unroll
            for (int r = 0; r < 8; ++r) {
                int gm = bm + wm + i * 16 + ro + r;
                int gn = bn + wn + j * 16 + cn;
                C[(size_t)gm * ldc + gn] = acc[i][j][r];
            }
}

// ---------------------------------------------------------------------------
// Kronecker mixer: per token t:
//   pa = silu(p[:4096]).reshape(64,64)/8 ; pb = silu(p[4096:]).reshape(32,32)/sqrt(32)
//   X = (x - y).reshape(64,32) ; out = pa @ X @ pb^T ; x_tilde = out + y
// One block (8 waves) per token; both small GEMMs done with WMMA via LDS.
// ---------------------------------------------------------------------------
__global__ __launch_bounds__(256) void kron_mixer_kernel(
    const float* __restrict__ p, const float* __restrict__ x,
    const float* __restrict__ y, float* __restrict__ xt)
{
    __shared__ __align__(64) bf16 pa_s[64 * 64];  // row-major 64x64 (M x K)
    __shared__ __align__(64) bf16 xT_s[32 * 64];  // X^T: [n][j], K contiguous per n
    __shared__ __align__(64) bf16 pb_s[32 * 32];  // row-major 32x32 (N x K)
    __shared__ __align__(64) bf16 tmp_s[64 * 32]; // row-major 64x32 (M x K for gemm2)

    const int tok = blockIdx.x;
    const int tid = threadIdx.x;
    const float* pr = p + (size_t)tok * PDIM;
    const float* xr = x + (size_t)tok * DI_SZ;
    const float* yr = y + (size_t)tok * DI_SZ;

    for (int i = tid; i < 4096; i += 256) pa_s[i] = f32_to_bf16(siluf(pr[i]) * 0.125f);
    const float is2 = 0.17677669529663687f; // 1/sqrt(32)
    for (int i = tid; i < 1024; i += 256) pb_s[i] = f32_to_bf16(siluf(pr[4096 + i]) * is2);
    for (int i = tid; i < 2048; i += 256) {
        int n = i >> 6, j = i & 63; // xT_s[n*64+j] = X[j][n]
        xT_s[i] = f32_to_bf16(xr[j * 32 + n] - yr[j * 32 + n]);
    }
    __syncthreads();

    const int lane = tid & 31, wave = tid >> 5;
    const int mt = wave >> 1, nt = wave & 1;
    const int m0 = mt * 16, n0 = nt * 16;

    // GEMM1: tmp(64x32) = pa(64x64) @ X(64x32)   (B operand = X^T rows, ldb=64)
    {
        f32x8 acc = {};
        for (int k0 = 0; k0 < 64; k0 += 32) {
            bf16x16 a = load_a_frag(pa_s, 64, m0, k0, lane);
            bf16x16 b = load_b_frag(xT_s, 64, n0, k0, lane);
            acc = wmma_bf16(a, b, acc);
        }
        const int cn = lane & 15, ro = (lane >> 4) * 8;
#pragma unroll
        for (int r = 0; r < 8; ++r)
            tmp_s[(m0 + ro + r) * 32 + n0 + cn] = f32_to_bf16(acc[r]);
    }
    __syncthreads();

    // GEMM2: out(64x32) = tmp(64x32) @ pb(32x32)^T ; write x_tilde = out + y
    {
        f32x8 acc = {};
        bf16x16 a = load_a_frag(tmp_s, 32, m0, 0, lane);
        bf16x16 b = load_b_frag(pb_s, 32, n0, 0, lane);
        acc = wmma_bf16(a, b, acc);
        const int cn = lane & 15, ro = (lane >> 4) * 8;
        float* xo = xt + (size_t)tok * DI_SZ;
#pragma unroll
        for (int r = 0; r < 8; ++r) {
            int idx = (m0 + ro + r) * 32 + n0 + cn;
            xo[idx] = acc[r] + yr[idx];
        }
    }
}

// ---------------------------------------------------------------------------
// Elementwise / normalization kernels
// ---------------------------------------------------------------------------
__global__ void cvt_f32_bf16_kernel(const float* __restrict__ s, bf16* __restrict__ d, size_t n) {
    size_t i = (size_t)blockIdx.x * blockDim.x + threadIdx.x;
    size_t st = (size_t)gridDim.x * blockDim.x;
    for (; i < n; i += st) d[i] = f32_to_bf16(s[i]);
}
__global__ void zero_f32_kernel(float* __restrict__ d, size_t n) {
    size_t i = (size_t)blockIdx.x * blockDim.x + threadIdx.x;
    size_t st = (size_t)gridDim.x * blockDim.x;
    for (; i < n; i += st) d[i] = 0.f;
}
__global__ void zero_bf16_kernel(bf16* __restrict__ d, size_t n) {
    size_t i = (size_t)blockIdx.x * blockDim.x + threadIdx.x;
    size_t st = (size_t)gridDim.x * blockDim.x;
    for (; i < n; i += st) d[i] = bf16_zero();
}

// row-normalized weight-norm weights, emitted as bf16: w = g * v / ||v||_row
__global__ __launch_bounds__(256) void wn_rownorm_bf16_kernel(
    const float* __restrict__ v, const float* __restrict__ g, bf16* __restrict__ w, int K)
{
    const int row = blockIdx.x, tid = threadIdx.x;
    const float* vr = v + (size_t)row * K;
    float ss = 0.f;
    for (int i = tid; i < K; i += 256) { float t = vr[i]; ss += t * t; }
    for (int o = 16; o > 0; o >>= 1) ss += __shfl_xor(ss, o, 32);
    __shared__ float red[8];
    if ((tid & 31) == 0) red[tid >> 5] = ss;
    __syncthreads();
    float tot = 0.f;
#pragma unroll
    for (int i = 0; i < 8; ++i) tot += red[i];
    const float scale = g[row] / sqrtf(tot);
    bf16* wr = w + (size_t)row * K;
    for (int i = tid; i < K; i += 256) wr[i] = f32_to_bf16(vr[i] * scale);
}

// split zxbcdt row (ld = DIP_PAD) into z, x=silu (f32 + bf16 into param_arg), Bv, Cv, dt
__global__ __launch_bounds__(256) void split_inproj_kernel(
    const float* __restrict__ zx, float* __restrict__ z, float* __restrict__ xf,
    bf16* __restrict__ parg, float* __restrict__ Bv, float* __restrict__ Cv,
    float* __restrict__ dtr)
{
    const int tok = blockIdx.x, tid = threadIdx.x;
    const float* r = zx + (size_t)tok * DIP_PAD;
    for (int i = tid; i < DI_SZ; i += 256) z[(size_t)tok * DI_SZ + i] = r[i];
    for (int i = tid; i < DI_SZ; i += 256) {
        float v = siluf(r[DI_SZ + i]);
        xf[(size_t)tok * DI_SZ + i] = v;
        parg[(size_t)tok * (2 * DI_SZ) + i] = f32_to_bf16(v);
    }
    for (int i = tid; i < DS_SZ; i += 256) {
        Bv[(size_t)tok * DS_SZ + i] = r[2 * DI_SZ + i];
        Cv[(size_t)tok * DS_SZ + i] = r[2 * DI_SZ + DS_SZ + i];
    }
    for (int i = tid; i < NH_SZ; i += 256) dtr[(size_t)tok * NH_SZ + i] = r[2 * DI_SZ + 2 * DS_SZ + i];
}

// ys = shift1(y) converted to bf16 into param_arg[:, DI:2*DI]
__global__ __launch_bounds__(256) void shift_y_bf16_kernel(
    const float* __restrict__ y, bf16* __restrict__ parg)
{
    const int tok = blockIdx.x, tid = threadIdx.x;
    bf16* d = parg + (size_t)tok * (2 * DI_SZ) + DI_SZ;
    if ((tok % L_SZ) == 0) {
        for (int i = tid; i < DI_SZ; i += 256) d[i] = bf16_zero();
    } else {
        const float* s = y + (size_t)(tok - 1) * DI_SZ;
        for (int i = tid; i < DI_SZ; i += 256) d[i] = f32_to_bf16(s[i]);
    }
}

// Bn = l2norm(Bv + BCy[:, :DS]) ; Cn = l2norm(Cv + BCy[:, DS:])
__global__ __launch_bounds__(256) void bc_combine_kernel(
    const float* __restrict__ Bv, const float* __restrict__ Cv,
    const float* __restrict__ BCy, float* __restrict__ Bn, float* __restrict__ Cn)
{
    const int tok = blockIdx.x, tid = threadIdx.x;
    const bool isC = tid >= 128;
    const int i = tid & 127;
    float v = (isC ? Cv : Bv)[(size_t)tok * DS_SZ + i]
            + BCy[(size_t)tok * (2 * DS_SZ) + (isC ? DS_SZ : 0) + i];
    float ss = v * v;
    for (int o = 16; o > 0; o >>= 1) ss += __shfl_xor(ss, o, 32);
    __shared__ float red[8];
    if ((tid & 31) == 0) red[tid >> 5] = ss;
    __syncthreads();
    float tot = isC ? (red[4] + red[5] + red[6] + red[7])
                    : (red[0] + red[1] + red[2] + red[3]);
    float out = v / fmaxf(sqrtf(tot), EPS_F);
    (isC ? Cn : Bn)[(size_t)tok * DS_SZ + i] = out;
}

// rmsnorm over DI with weight
__global__ __launch_bounds__(256) void rmsnorm_kernel(
    const float* __restrict__ yin, const float* __restrict__ w, float* __restrict__ yout)
{
    const int tok = blockIdx.x, tid = threadIdx.x;
    const float* r = yin + (size_t)tok * DI_SZ;
    float ss = 0.f;
    for (int i = tid; i < DI_SZ; i += 256) { float t = r[i]; ss += t * t; }
    for (int o = 16; o > 0; o >>= 1) ss += __shfl_xor(ss, o, 32);
    __shared__ float red[8];
    if ((tid & 31) == 0) red[tid >> 5] = ss;
    __syncthreads();
    float tot = 0.f;
#pragma unroll
    for (int i = 0; i < 8; ++i) tot += red[i];
    const float inv = rsqrtf(tot / (float)DI_SZ + EPS_F);
    float* o = yout + (size_t)tok * DI_SZ;
    for (int i = tid; i < DI_SZ; i += 256) o[i] = r[i] * inv * w[i];
}

// g = silu(z + z2tmp) * y  ->  bf16
__global__ __launch_bounds__(256) void final_gate_kernel(
    const float* __restrict__ z, const float* __restrict__ t,
    const float* __restrict__ y, bf16* __restrict__ g)
{
    const int tok = blockIdx.x, tid = threadIdx.x;
    for (int i = tid; i < DI_SZ; i += 256) {
        size_t idx = (size_t)tok * DI_SZ + i;
        g[idx] = f32_to_bf16(siluf(z[idx] + t[idx]) * y[idx]);
    }
}

// ---------------------------------------------------------------------------
// SSD scan: one block per (batch, head). State s[HP=64][DS=128] f32 lives in
// registers: thread t owns p = t>>2, n in [(t&3)*32, +32). Sequential over L.
// ---------------------------------------------------------------------------
__global__ __launch_bounds__(256) void ssd_scan_kernel(
    const float* __restrict__ xt,  const float* __restrict__ dtr,
    const float* __restrict__ Bn,  const float* __restrict__ Cn,
    const float* __restrict__ A_log, const float* __restrict__ dt_bias,
    const float* __restrict__ Dp,  float* __restrict__ yout)
{
    const int bh = blockIdx.x;           // 0..63
    const int b = bh >> 5, h = bh & 31;
    const int tid = threadIdx.x;
    const int pidx = tid >> 2;           // 0..63
    const int nb = (tid & 3) * 32;       // n base
    const float aexp = __expf(A_log[h]);
    const float bias = dt_bias[h];
    const float Dh   = Dp[h];

    float s[32];
#pragma unroll
    for (int i = 0; i < 32; ++i) s[i] = 0.f;

    __shared__ float Bs[DS_SZ], Cs[DS_SZ], xs[HP_SZ];
    for (int l = 0; l < L_SZ; ++l) {
        const int tok = b * L_SZ + l;
        if (tid < 128)      Bs[tid]       = Bn[(size_t)tok * DS_SZ + tid];
        else                Cs[tid - 128] = Cn[(size_t)tok * DS_SZ + (tid - 128)];
        if (tid < 64)       xs[tid]       = xt[(size_t)tok * DI_SZ + h * HP_SZ + tid];
        __syncthreads();

        float dtv = dtr[(size_t)tok * NH_SZ + h] + bias;
        dtv = (dtv > 20.f) ? dtv : log1pf(__expf(dtv));   // softplus
        const float dA  = __expf(-aexp * dtv);
        const float xh  = xs[pidx];
        const float dtx = dtv * xh;

        float part = 0.f;
#pragma unroll
        for (int i = 0; i < 32; ++i) {
            s[i] = dA * s[i] + dtx * Bs[nb + i];
            part += s[i] * Cs[nb + i];
        }
        part += __shfl_xor(part, 1, 32);
        part += __shfl_xor(part, 2, 32);
        if ((tid & 3) == 0)
            yout[(size_t)tok * DI_SZ + h * HP_SZ + pidx] = part + Dh * xh;
        __syncthreads();
    }
}

// ---------------------------------------------------------------------------
// Host-side orchestration
// ---------------------------------------------------------------------------
static inline size_t align256(size_t x) { return (x + 255) & ~(size_t)255; }

static void launch_gemm(const bf16* A, const bf16* B, float* C,
                        int M, int N, int K, int lda, int ldb, int ldc,
                        hipStream_t s)
{
    int blocks = (M / BM) * (N / BN);
    hipLaunchKernelGGL(gemm_bf16_lds, dim3(blocks), dim3(256), 0, s,
                       A, B, C, M, N, K, lda, ldb, ldc);
}

extern "C" void kernel_launch(void* const* d_in, const int* in_sizes, int n_in,
                              void* d_out, int out_size, void* d_ws, size_t ws_size,
                              hipStream_t stream)
{
    (void)in_sizes; (void)n_in; (void)out_size; (void)ws_size;
    const float* u       = (const float*)d_in[0];
    const float* W_in    = (const float*)d_in[1];
    const float* W_mix   = (const float*)d_in[2];
    const float* bc_v    = (const float*)d_in[3];
    const float* bc_g    = (const float*)d_in[4];
    const float* zy_v    = (const float*)d_in[5];
    const float* zy_g    = (const float*)d_in[6];
    const float* dt_bias = (const float*)d_in[7];
    const float* A_log   = (const float*)d_in[8];
    const float* Dp      = (const float*)d_in[9];
    const float* norm_w  = (const float*)d_in[10];
    const float* W_out   = (const float*)d_in[11];
    float* out = (float*)d_out;

    // ---- workspace carve-out -------------------------------------------------
    char* ws = (char*)d_ws;
    size_t off = 0;
    auto carve = [&](size_t bytes) { void* p = ws + off; off += align256(bytes); return p; };

    bf16*  u_bf    = (bf16*) carve((size_t)TOK * DM_SZ * sizeof(bf16));
    bf16*  Win_bf  = (bf16*) carve((size_t)DIP_PAD * DM_SZ * sizeof(bf16));
    bf16*  Wmix_bf = (bf16*) carve((size_t)PDIM * (2 * DI_SZ) * sizeof(bf16));
    bf16*  Wout_bf = (bf16*) carve((size_t)DM_SZ * DI_SZ * sizeof(bf16));
    bf16*  bcw_bf  = (bf16*) carve((size_t)(2 * DS_SZ) * DI_SZ * sizeof(bf16));
    bf16*  zyw_bf  = (bf16*) carve((size_t)DI_SZ * DI_SZ * sizeof(bf16));
    float* zxbcdt  = (float*)carve((size_t)TOK * DIP_PAD * sizeof(float));
    float* z_buf   = (float*)carve((size_t)TOK * DI_SZ * sizeof(float));
    float* x_buf   = (float*)carve((size_t)TOK * DI_SZ * sizeof(float));
    bf16*  parg_bf = (bf16*) carve((size_t)TOK * (2 * DI_SZ) * sizeof(bf16));
    float* Bv_buf  = (float*)carve((size_t)TOK * DS_SZ * sizeof(float));
    float* Cv_buf  = (float*)carve((size_t)TOK * DS_SZ * sizeof(float));
    float* dt_buf  = (float*)carve((size_t)TOK * NH_SZ * sizeof(float));
    float* p_buf   = (float*)carve((size_t)TOK * PDIM * sizeof(float));
    float* y_buf   = (float*)carve((size_t)TOK * DI_SZ * sizeof(float));
    float* ys_buf  = (float*)carve((size_t)TOK * DI_SZ * sizeof(float)); // raw scan out
    float* xt_buf  = (float*)carve((size_t)TOK * DI_SZ * sizeof(float));
    float* Bn_buf  = (float*)carve((size_t)TOK * DS_SZ * sizeof(float));
    float* Cn_buf  = (float*)carve((size_t)TOK * DS_SZ * sizeof(float));
    float* BCy_buf = (float*)carve((size_t)TOK * (2 * DS_SZ) * sizeof(float));
    float* z2_buf  = (float*)carve((size_t)TOK * DI_SZ * sizeof(float));
    bf16*  g_bf    = (bf16*) carve((size_t)TOK * DI_SZ * sizeof(bf16));

    // ---- weight prep ---------------------------------------------------------
    hipLaunchKernelGGL(cvt_f32_bf16_kernel, dim3(2048), dim3(256), 0, stream,
                       u, u_bf, (size_t)TOK * DM_SZ);
    hipLaunchKernelGGL(cvt_f32_bf16_kernel, dim3(2048), dim3(256), 0, stream,
                       W_in, Win_bf, (size_t)DIP * DM_SZ);
    hipLaunchKernelGGL(zero_bf16_kernel, dim3(256), dim3(256), 0, stream,
                       Win_bf + (size_t)DIP * DM_SZ,
                       (size_t)(DIP_PAD - DIP) * DM_SZ);   // zero pad rows
    hipLaunchKernelGGL(cvt_f32_bf16_kernel, dim3(4096), dim3(256), 0, stream,
                       W_mix, Wmix_bf, (size_t)PDIM * (2 * DI_SZ));
    hipLaunchKernelGGL(cvt_f32_bf16_kernel, dim3(2048), dim3(256), 0, stream,
                       W_out, Wout_bf, (size_t)DM_SZ * DI_SZ);
    hipLaunchKernelGGL(wn_rownorm_bf16_kernel, dim3(2 * DS_SZ), dim3(256), 0, stream,
                       bc_v, bc_g, bcw_bf, DI_SZ);
    hipLaunchKernelGGL(wn_rownorm_bf16_kernel, dim3(DI_SZ), dim3(256), 0, stream,
                       zy_v, zy_g, zyw_bf, DI_SZ);

    // ---- in-projection: zxbcdt = u @ W_in^T (N padded to 4480) --------------
    launch_gemm(u_bf, Win_bf, zxbcdt, TOK, DIP_PAD, DM_SZ, DM_SZ, DM_SZ, DIP_PAD, stream);
    hipLaunchKernelGGL(split_inproj_kernel, dim3(TOK), dim3(256), 0, stream,
                       zxbcdt, z_buf, x_buf, parg_bf, Bv_buf, Cv_buf, dt_buf);
    hipLaunchKernelGGL(zero_f32_kernel, dim3(2048), dim3(256), 0, stream,
                       y_buf, (size_t)TOK * DI_SZ);

    // ---- fixed-point iterations (3 fp + 1 bw, numerically identical) --------
    for (int it = 0; it < N_ITER; ++it) {
        hipLaunchKernelGGL(shift_y_bf16_kernel, dim3(TOK), dim3(256), 0, stream,
                           y_buf, parg_bf);
        // p = param_arg @ W_mix^T  (4096 x 5120 x 4096)
        launch_gemm(parg_bf, Wmix_bf, p_buf, TOK, PDIM, 2 * DI_SZ,
                    2 * DI_SZ, 2 * DI_SZ, PDIM, stream);
        // x_tilde = kron(pa, pb) applied to (x - y), + y
        hipLaunchKernelGGL(kron_mixer_kernel, dim3(TOK), dim3(256), 0, stream,
                           p_buf, x_buf, y_buf, xt_buf);
        // BCy = ys @ bc_w^T  (A = second half of param_arg, lda = 2*DI)
        launch_gemm(parg_bf + DI_SZ, bcw_bf, BCy_buf, TOK, 2 * DS_SZ, DI_SZ,
                    2 * DI_SZ, DI_SZ, 2 * DS_SZ, stream);
        hipLaunchKernelGGL(bc_combine_kernel, dim3(TOK), dim3(256), 0, stream,
                           Bv_buf, Cv_buf, BCy_buf, Bn_buf, Cn_buf);
        // sequential SSD scan
        hipLaunchKernelGGL(ssd_scan_kernel, dim3(B_SZ * NH_SZ), dim3(256), 0, stream,
                           xt_buf, dt_buf, Bn_buf, Cn_buf, A_log, dt_bias, Dp, ys_buf);
        hipLaunchKernelGGL(rmsnorm_kernel, dim3(TOK), dim3(256), 0, stream,
                           ys_buf, norm_w, y_buf);
    }

    // ---- output head ---------------------------------------------------------
    hipLaunchKernelGGL(shift_y_bf16_kernel, dim3(TOK), dim3(256), 0, stream,
                       y_buf, parg_bf);
    // z2tmp = shift1(y) @ zy_w^T
    launch_gemm(parg_bf + DI_SZ, zyw_bf, z2_buf, TOK, DI_SZ, DI_SZ,
                2 * DI_SZ, DI_SZ, DI_SZ, stream);
    hipLaunchKernelGGL(final_gate_kernel, dim3(TOK), dim3(256), 0, stream,
                       z_buf, z2_buf, y_buf, g_bf);
    // out = g @ W_out^T
    launch_gemm(g_bf, Wout_bf, out, TOK, DM_SZ, DI_SZ, DI_SZ, DI_SZ, DM_SZ, stream);
}
